// SelfAttention_16733192585752
// MI455X (gfx1250) — matrix operand order repaired and checked by reading, not compiled, not run on hardware
//
#include <hip/hip_runtime.h>

typedef _Float16 half_t;
typedef __attribute__((ext_vector_type(16))) _Float16 v16h;
typedef __attribute__((ext_vector_type(8)))  _Float16 v8h;
typedef __attribute__((ext_vector_type(8)))  float    v8f;
typedef __attribute__((ext_vector_type(4)))  unsigned int v4u;
typedef __attribute__((ext_vector_type(8)))  int v8i;
typedef __attribute__((ext_vector_type(4)))  int v4i;

#define DMODEL 512
#define T_SEQ  2048
#define NBATCH 4
#define NHEAD  8
#define DHEAD  64
#define TOKENS (NBATCH * T_SEQ)   // 8192

__device__ __forceinline__ int lane_id() { return (int)(threadIdx.x & 31); }

// A-fragment (16x32 f16, MxK): lane m = lane&15; kb = lane<16 ? 0 : 8.
__device__ __forceinline__ v16h load_a_frag(const half_t* a, int ld) {
  const int lane = lane_id();
  const int m  = lane & 15;
  const int kb = (lane & 16) ? 8 : 0;
  const half_t* p = a + m * ld + kb;
  union { v16h v; v8h h[2]; } u;
  u.h[0] = *(const v8h*)(p);
  u.h[1] = *(const v8h*)(p + 16);
  return u.v;
}

// B-fragment (32x16 f16, KxN) from B^T stored row-major [N][K].
__device__ __forceinline__ v16h load_b_frag(const half_t* bt, int ld) {
  const int lane = lane_id();
  const int n  = lane & 15;
  const int kb = (lane & 16) ? 8 : 0;
  const half_t* p = bt + n * ld + kb;
  union { v16h v; v8h h[2]; } u;
  u.h[0] = *(const v8h*)(p);
  u.h[1] = *(const v8h*)(p + 16);
  return u.v;
}

__device__ __forceinline__ v8f wmma_f16(v16h a, v16h b, v8f c) {
  return __builtin_amdgcn_wmma_f32_16x16x32_f16(false, a, false, b, (short)0, c,
                                                false, false);
}

// TDM: DMA a 2D f16 tile [dim1 rows][dim0 elems] (row stride = stride0 elems)
// from global memory to LDS (packed row-major, ld = dim0). ISA ch. 8 D#:
// group0 = count=1 | lds_addr | 57b global addr | type=2 (bits 127:126).
// group1 = data_size=1(2B) | tensor dims == tile dims (no OOB) | stride0.
// This toolchain exposes the 6-arg builtin (extra int32x8 group before cpol).
__device__ __forceinline__ void tdm_load_2d_f16(const half_t* gsrc,
                                                half_t* lds_dst,
                                                unsigned dim0, unsigned dim1,
                                                unsigned stride0) {
  const unsigned long long ga = (unsigned long long)(uintptr_t)gsrc;
  const unsigned lds_off = (unsigned)(uintptr_t)lds_dst;  // low 32 = LDS byte addr
  v4u g0;
  g0[0] = 1u;                                              // count = 1 valid D#
  g0[1] = lds_off;
  g0[2] = (unsigned)ga;
  g0[3] = ((unsigned)(ga >> 32) & 0x01FFFFFFu) | (2u << 30); // type = 2 (image)
  v8i g1;
  g1[0] = (int)(1u << 16);                                 // data_size = 2 bytes
  g1[1] = (int)((dim0 & 0xFFFFu) << 16);                   // tensor_dim0[15:0]
  g1[2] = (int)((dim0 >> 16) | ((dim1 & 0xFFFFu) << 16));  // dim0 hi | dim1 lo
  g1[3] = (int)((dim1 >> 16) | ((dim0 & 0xFFFFu) << 16));  // dim1 hi | tile_dim0
  g1[4] = (int)(dim1 & 0xFFFFu);                           // tile_dim1, tile_dim2=0
  g1[5] = (int)stride0;                                    // tensor_dim0_stride lo
  g1[6] = 0;                                               // stride hi | stride1 lo
  g1[7] = 0;
  const v4i z4 = {0, 0, 0, 0};                             // 2D: groups 2/3 unused
  const v8i z8 = {0, 0, 0, 0, 0, 0, 0, 0};
  __builtin_amdgcn_tensor_load_to_lds(g0, g1, z4, z4, z8, 0);
}

// ---------------------------------------------------------------- converts
__global__ void cvt_f32_to_f16(const float* __restrict__ in,
                               half_t* __restrict__ out, int n) {
  int i = blockIdx.x * 256 + threadIdx.x;
  if (i < n) out[i] = (half_t)in[i];
}

// Vh [TOKENS][DMODEL] -> Vt [B][H][DHEAD][T_SEQ]
__global__ void transpose_v(const half_t* __restrict__ Vh,
                            half_t* __restrict__ Vt) {
  int i = blockIdx.x * 256 + threadIdx.x;
  if (i >= TOKENS * DMODEL) return;
  int c   = i & (DMODEL - 1);
  int tok = i >> 9;
  int b = tok >> 11;
  int t = tok & (T_SEQ - 1);
  int h = c >> 6;
  int s = c & (DHEAD - 1);
  Vt[(((size_t)(b * NHEAD + h) * DHEAD) + s) * T_SEQ + t] = Vh[i];
}

// ---------------------------------------------------------------- projections
// C[M][512] = A[M][512] @ W[512(out)][512(in)]^T.
// One wave: 32x64 tile, software-pipelined K loop. grid = (M/32, 512/256).
__global__ void __launch_bounds__(128) gemm_qkv(const half_t* __restrict__ A,
                                                const half_t* __restrict__ W,
                                                half_t* __restrict__ C,
                                                float scale) {
  const int wid = threadIdx.x >> 5;
  const int m0 = blockIdx.x * 32;
  const int n0 = blockIdx.y * 256 + wid * 64;
  const half_t* A0 = A + (size_t)m0 * DMODEL;
  const half_t* A1 = A0 + 16 * DMODEL;
  const half_t* W0 = W + (size_t)n0 * DMODEL;

  v8f c0 = {}, c1 = {}, c2 = {}, c3 = {}, c4 = {}, c5 = {}, c6 = {}, c7 = {};

  auto mm = [&](v16h fa0, v16h fa1, v16h fb0, v16h fb1, v16h fb2, v16h fb3) {
    c0 = wmma_f16(fa0, fb0, c0); c1 = wmma_f16(fa0, fb1, c1);
    c2 = wmma_f16(fa0, fb2, c2); c3 = wmma_f16(fa0, fb3, c3);
    c4 = wmma_f16(fa1, fb0, c4); c5 = wmma_f16(fa1, fb1, c5);
    c6 = wmma_f16(fa1, fb2, c6); c7 = wmma_f16(fa1, fb3, c7);
  };

  v16h a0 = load_a_frag(A0, DMODEL);
  v16h a1 = load_a_frag(A1, DMODEL);
  v16h b0 = load_b_frag(W0,               DMODEL);
  v16h b1 = load_b_frag(W0 + 16 * DMODEL, DMODEL);
  v16h b2 = load_b_frag(W0 + 32 * DMODEL, DMODEL);
  v16h b3 = load_b_frag(W0 + 48 * DMODEL, DMODEL);

  #pragma unroll 2
  for (int k = 32; k < DMODEL; k += 32) {
    v16h na0 = load_a_frag(A0 + k, DMODEL);
    v16h na1 = load_a_frag(A1 + k, DMODEL);
    v16h nb0 = load_b_frag(W0 + k,               DMODEL);
    v16h nb1 = load_b_frag(W0 + 16 * DMODEL + k, DMODEL);
    v16h nb2 = load_b_frag(W0 + 32 * DMODEL + k, DMODEL);
    v16h nb3 = load_b_frag(W0 + 48 * DMODEL + k, DMODEL);
    mm(a0, a1, b0, b1, b2, b3);
    a0 = na0; a1 = na1; b0 = nb0; b1 = nb1; b2 = nb2; b3 = nb3;
  }
  mm(a0, a1, b0, b1, b2, b3);

  const int lane = lane_id();
  const int n  = lane & 15;
  const int r0 = (lane & 16) ? 8 : 0;
  v8f* accs[2][4] = {{&c0, &c1, &c2, &c3}, {&c4, &c5, &c6, &c7}};
  #pragma unroll
  for (int mh = 0; mh < 2; ++mh) {
    #pragma unroll
    for (int r = 0; r < 8; ++r) {
      half_t* crow = C + (size_t)(m0 + mh * 16 + r0 + r) * DMODEL + n0 + n;
      crow[0]  = (half_t)((*accs[mh][0])[r] * scale);
      crow[16] = (half_t)((*accs[mh][1])[r] * scale);
      crow[32] = (half_t)((*accs[mh][2])[r] * scale);
      crow[48] = (half_t)((*accs[mh][3])[r] * scale);
    }
  }
}

// Output projection: f32 result + bias into d_out.
__global__ void __launch_bounds__(128) gemm_out(const half_t* __restrict__ A,
                                                const half_t* __restrict__ W,
                                                const float* __restrict__ bias,
                                                float* __restrict__ C) {
  const int wid = threadIdx.x >> 5;
  const int m0 = blockIdx.x * 32;
  const int n0 = blockIdx.y * 256 + wid * 64;
  const half_t* A0 = A + (size_t)m0 * DMODEL;
  const half_t* A1 = A0 + 16 * DMODEL;
  const half_t* W0 = W + (size_t)n0 * DMODEL;

  v8f c0 = {}, c1 = {}, c2 = {}, c3 = {}, c4 = {}, c5 = {}, c6 = {}, c7 = {};

  auto mm = [&](v16h fa0, v16h fa1, v16h fb0, v16h fb1, v16h fb2, v16h fb3) {
    c0 = wmma_f16(fa0, fb0, c0); c1 = wmma_f16(fa0, fb1, c1);
    c2 = wmma_f16(fa0, fb2, c2); c3 = wmma_f16(fa0, fb3, c3);
    c4 = wmma_f16(fa1, fb0, c4); c5 = wmma_f16(fa1, fb1, c5);
    c6 = wmma_f16(fa1, fb2, c6); c7 = wmma_f16(fa1, fb3, c7);
  };

  v16h a0 = load_a_frag(A0, DMODEL);
  v16h a1 = load_a_frag(A1, DMODEL);
  v16h b0 = load_b_frag(W0,               DMODEL);
  v16h b1 = load_b_frag(W0 + 16 * DMODEL, DMODEL);
  v16h b2 = load_b_frag(W0 + 32 * DMODEL, DMODEL);
  v16h b3 = load_b_frag(W0 + 48 * DMODEL, DMODEL);

  #pragma unroll 2
  for (int k = 32; k < DMODEL; k += 32) {
    v16h na0 = load_a_frag(A0 + k, DMODEL);
    v16h na1 = load_a_frag(A1 + k, DMODEL);
    v16h nb0 = load_b_frag(W0 + k,               DMODEL);
    v16h nb1 = load_b_frag(W0 + 16 * DMODEL + k, DMODEL);
    v16h nb2 = load_b_frag(W0 + 32 * DMODEL + k, DMODEL);
    v16h nb3 = load_b_frag(W0 + 48 * DMODEL + k, DMODEL);
    mm(a0, a1, b0, b1, b2, b3);
    a0 = na0; a1 = na1; b0 = nb0; b1 = nb1; b2 = nb2; b3 = nb3;
  }
  mm(a0, a1, b0, b1, b2, b3);

  const int lane = lane_id();
  const int n  = lane & 15;
  const int r0 = (lane & 16) ? 8 : 0;
  v8f* accs[2][4] = {{&c0, &c1, &c2, &c3}, {&c4, &c5, &c6, &c7}};
  #pragma unroll
  for (int mh = 0; mh < 2; ++mh) {
    #pragma unroll
    for (int r = 0; r < 8; ++r) {
      float* crow = C + (size_t)(m0 + mh * 16 + r0 + r) * DMODEL + n0 + n;
      crow[0]  = (*accs[mh][0])[r] + bias[n0 + n];
      crow[16] = (*accs[mh][1])[r] + bias[n0 + 16 + n];
      crow[32] = (*accs[mh][2])[r] + bias[n0 + 32 + n];
      crow[48] = (*accs[mh][3])[r] + bias[n0 + 48 + n];
    }
  }
}

// ---------------------------------------------------------------- attention
// Block = 4 waves sharing one (b,h); each wave owns a 16-query block.
// K/V chunks (32 keys) are DMA'd into double-buffered LDS by the Tensor Data
// Mover (issued by wave 0, TENSORcnt-tracked), overlapping the WMMA pipeline;
// all 4 waves consume the shared LDS tiles. Softmax scale folded into Q.
__global__ void __launch_bounds__(128) attn_fused(const half_t* __restrict__ Qh,
                                                  const half_t* __restrict__ Kh,
                                                  const half_t* __restrict__ Vt,
                                                  half_t* __restrict__ Oh) {
  __shared__ __align__(16) half_t Klds[2][32 * DHEAD];  // [keys][dims] ld=64
  __shared__ __align__(16) half_t Vlds[2][DHEAD * 32];  // [dims][keys] ld=32
  __shared__ __align__(16) half_t Plds[4][16 * 32];
  const int wid = threadIdx.x >> 5;
  const int qb  = blockIdx.x * 4 + wid;
  const int h   = blockIdx.y;
  const int b   = blockIdx.z;
  const int lane = lane_id();
  const int n  = lane & 15;
  const int r0 = (lane & 16) ? 8 : 0;

  const half_t* Kbh = Kh + (size_t)b * T_SEQ * DMODEL + h * DHEAD;
  const half_t* Vbh = Vt + (size_t)(b * NHEAD + h) * DHEAD * T_SEQ;

  const half_t* Qbase = Qh + (size_t)(b * T_SEQ + qb * 16) * DMODEL + h * DHEAD;
  const v16h qa0 = load_a_frag(Qbase +  0, DMODEL);
  const v16h qa1 = load_a_frag(Qbase + 32, DMODEL);

  v8f o0 = {}, o1 = {}, o2 = {}, o3 = {};
  float m_run[8], l_run[8];
  #pragma unroll
  for (int r = 0; r < 8; ++r) { m_run[r] = -1e30f; l_run[r] = 0.f; }

  half_t* Pw = Plds[wid];

  // prologue: DMA chunk 0 into buffer 0
  if (wid == 0) {
    tdm_load_2d_f16(Kbh, Klds[0], DHEAD, 32, DMODEL);  // 32 key rows of 64 f16
    tdm_load_2d_f16(Vbh, Vlds[0], 32, DHEAD, T_SEQ);   // 64 dim rows of 32 f16
    __builtin_amdgcn_s_wait_tensorcnt(0);
  }
  __syncthreads();

  int buf = 0;
  for (int j0 = 0; j0 < T_SEQ; j0 += 32, buf ^= 1) {
    // kick off next chunk's DMA into the other buffer (overlaps compute)
    if (wid == 0 && j0 + 32 < T_SEQ) {
      tdm_load_2d_f16(Kbh + (size_t)(j0 + 32) * DMODEL, Klds[buf ^ 1],
                      DHEAD, 32, DMODEL);
      tdm_load_2d_f16(Vbh + (j0 + 32), Vlds[buf ^ 1], 32, DHEAD, T_SEQ);
    }

    const half_t* Kl = Klds[buf];
    const half_t* Vl = Vlds[buf];

    v8f s0 = {}, s1 = {};
    s0 = wmma_f16(qa0, load_b_frag(Kl,                DHEAD), s0);
    s0 = wmma_f16(qa1, load_b_frag(Kl + 32,           DHEAD), s0);
    s1 = wmma_f16(qa0, load_b_frag(Kl + 16 * DHEAD,      DHEAD), s1);
    s1 = wmma_f16(qa1, load_b_frag(Kl + 16 * DHEAD + 32, DHEAD), s1);

    // online softmax: a row lives across a 16-lane half at fixed vgpr slot
    float alpha[8];
    #pragma unroll
    for (int r = 0; r < 8; ++r) {
      float mj = fmaxf(s0[r], s1[r]);
      mj = fmaxf(mj, __shfl_xor(mj, 1, 32));
      mj = fmaxf(mj, __shfl_xor(mj, 2, 32));
      mj = fmaxf(mj, __shfl_xor(mj, 4, 32));
      mj = fmaxf(mj, __shfl_xor(mj, 8, 32));
      const float nm = fmaxf(m_run[r], mj);
      alpha[r] = __expf(m_run[r] - nm);
      m_run[r] = nm;
      const float p0 = __expf(s0[r] - nm);
      const float p1 = __expf(s1[r] - nm);
      s0[r] = p0; s1[r] = p1;
      float t = p0 + p1;
      t += __shfl_xor(t, 1, 32);
      t += __shfl_xor(t, 2, 32);
      t += __shfl_xor(t, 4, 32);
      t += __shfl_xor(t, 8, 32);
      l_run[r] = l_run[r] * alpha[r] + t;
    }
    #pragma unroll
    for (int r = 0; r < 8; ++r) {
      o0[r] *= alpha[r]; o1[r] *= alpha[r];
      o2[r] *= alpha[r]; o3[r] *= alpha[r];
    }

    // C-frag (col-per-lane) -> A-frag (row-per-lane) transpose via LDS
    #pragma unroll
    for (int r = 0; r < 8; ++r) {
      Pw[(r0 + r) * 32 + n]      = (half_t)s0[r];
      Pw[(r0 + r) * 32 + 16 + n] = (half_t)s1[r];
    }
    __syncthreads();
    const v16h pa = load_a_frag(Pw, 32);

    o0 = wmma_f16(pa, load_b_frag(Vl,           32), o0);
    o1 = wmma_f16(pa, load_b_frag(Vl + 16 * 32, 32), o1);
    o2 = wmma_f16(pa, load_b_frag(Vl + 32 * 32, 32), o2);
    o3 = wmma_f16(pa, load_b_frag(Vl + 48 * 32, 32), o3);

    // publish next chunk: DMA must have landed, all waves done with this one
    if (wid == 0 && j0 + 32 < T_SEQ) __builtin_amdgcn_s_wait_tensorcnt(0);
    __syncthreads();
  }

  // normalize and emit f16 for the output projection
  half_t* Orow = Oh + (size_t)(b * T_SEQ + qb * 16) * DMODEL + h * DHEAD;
  #pragma unroll
  for (int r = 0; r < 8; ++r) {
    const float inv = 1.0f / l_run[r];
    Orow[(size_t)(r0 + r) * DMODEL + n]      = (half_t)(o0[r] * inv);
    Orow[(size_t)(r0 + r) * DMODEL + 16 + n] = (half_t)(o1[r] * inv);
    Orow[(size_t)(r0 + r) * DMODEL + 32 + n] = (half_t)(o2[r] * inv);
    Orow[(size_t)(r0 + r) * DMODEL + 48 + n] = (half_t)(o3[r] * inv);
  }
}

// ---------------------------------------------------------------- launcher
extern "C" void kernel_launch(void* const* d_in, const int* in_sizes, int n_in,
                              void* d_out, int out_size, void* d_ws, size_t ws_size,
                              hipStream_t stream) {
  (void)in_sizes; (void)n_in; (void)out_size; (void)ws_size;
  const float* x  = (const float*)d_in[0];
  const float* Wq = (const float*)d_in[1];
  const float* Wk = (const float*)d_in[2];
  const float* Wv = (const float*)d_in[3];
  const float* Wo = (const float*)d_in[4];
  const float* bo = (const float*)d_in[5];
  float* out = (float*)d_out;

  const size_t NX = (size_t)TOKENS * DMODEL;   // 4,194,304
  const size_t NW = (size_t)DMODEL * DMODEL;   //   262,144

  half_t* w  = (half_t*)d_ws;
  half_t* xh  = w;
  half_t* Qh  = xh  + NX;
  half_t* Kh  = Qh  + NX;
  half_t* Vh  = Kh  + NX;      // reused as Oh after transpose_v
  half_t* Vt  = Vh  + NX;      // [B][H][DHEAD][T_SEQ]
  half_t* Wqh = Vt  + NX;
  half_t* Wkh = Wqh + NW;
  half_t* Wvh = Wkh + NW;
  half_t* Woh = Wvh + NW;
  half_t* Oh  = Vh;            // alias: Vh dead after transpose_v

  const int cvt_big = (int)((NX + 255) / 256);
  const int cvt_w   = (int)((NW + 255) / 256);
  cvt_f32_to_f16<<<cvt_big, 256, 0, stream>>>(x,  xh,  (int)NX);
  cvt_f32_to_f16<<<cvt_w,   256, 0, stream>>>(Wq, Wqh, (int)NW);
  cvt_f32_to_f16<<<cvt_w,   256, 0, stream>>>(Wk, Wkh, (int)NW);
  cvt_f32_to_f16<<<cvt_w,   256, 0, stream>>>(Wv, Wvh, (int)NW);
  cvt_f32_to_f16<<<cvt_w,   256, 0, stream>>>(Wo, Woh, (int)NW);

  const dim3 ggrid(TOKENS / 32, DMODEL / 256);
  const float qscale = 1.0f / sqrtf((float)DMODEL);
  gemm_qkv<<<ggrid, 128, 0, stream>>>(xh, Wqh, Qh, qscale);
  gemm_qkv<<<ggrid, 128, 0, stream>>>(xh, Wkh, Kh, 1.0f);
  gemm_qkv<<<ggrid, 128, 0, stream>>>(xh, Wvh, Vh, 1.0f);

  transpose_v<<<cvt_big, 256, 0, stream>>>(Vh, Vt);

  attn_fused<<<dim3(T_SEQ / 64, NHEAD, NBATCH), 128, 0, stream>>>(Qh, Kh, Vt, Oh);

  gemm_out<<<ggrid, 128, 0, stream>>>(Oh, Woh, bo, out);
}
